// Block_31722628448427
// MI455X (gfx1250) — compile-verified, matrix operand order checked by reading, not run on hardware
//
#include <hip/hip_runtime.h>
#include <hip/hip_bf16.h>
#include <stdint.h>

// ---------------------------------------------------------------------------
// Problem constants: B=4, S=4096, E=1024, H=16, HD=64, LD=256, NL=64, FF=4096.
// ---------------------------------------------------------------------------
typedef __attribute__((ext_vector_type(16))) __bf16 v16bf;
typedef __attribute__((ext_vector_type(8)))  float  v8f;
typedef int v4i __attribute__((vector_size(16)));

__device__ __forceinline__ unsigned short f2bf(float f) {
    union { float f; uint32_t u; } c{f};
    uint32_t u = c.u;
    uint32_t r = u + 0x7FFFu + ((u >> 16) & 1u);   // round-to-nearest-even
    return (unsigned short)(r >> 16);
}

union Frag {            // 16 bf16 = one WMMA 16-bit A/B fragment (8 VGPRs)
    uint4 q[2];
    v16bf v;
};

// ---------------------------------------------------------------------------
// gfx1250 async global->LDS copy (ASYNCcnt path), with sync fallback.
// Builtin signature (from probe): (AS1 v4i* src, AS3 v4i* dst, imm off, imm cpol)
// ---------------------------------------------------------------------------
#if __has_builtin(__builtin_amdgcn_global_load_async_to_lds_b128)
#define HAVE_ASYNC_LDS 1
__device__ __forceinline__ void cp16_g2l(const unsigned short* g, unsigned short* l) {
    unsigned short* gnc = const_cast<unsigned short*>(g);
    __builtin_amdgcn_global_load_async_to_lds_b128(
        (__attribute__((address_space(1))) v4i*)gnc,
        (__attribute__((address_space(3))) v4i*)l, 0, 0);
}
#else
#define HAVE_ASYNC_LDS 0
__device__ __forceinline__ void cp16_g2l(const unsigned short* g, unsigned short* l) {
    *(uint4*)l = *(const uint4*)g;
}
#endif

__device__ __forceinline__ void wait_async_lds() {
#if HAVE_ASYNC_LDS
#if __has_builtin(__builtin_amdgcn_s_wait_asynccnt)
    __builtin_amdgcn_s_wait_asynccnt(0);
#else
    asm volatile("s_wait_asynccnt 0x0" ::: "memory");
#endif
#endif
}

// ---------------------------------------------------------------------------
// Generic bf16 WMMA GEMM, double-buffered LDS, async staging.
// Workgroup = WM x WN waves, each wave owns a 32x32 C tile
// (2x2 v_wmma_f32_16x16x32_bf16 accumulators). Macro tile = 32WM x 32WN.
// A: [M][K] bf16 row-major (lda).  BT=true: B given as Bt[N][K] (ldb, async);
// BT=false: B is [K][N] (ldb), transposed while staging (sync path).
// Batched via blockIdx.z: offset = (z/zmod)*Sdiv + (z%zmod)*Smod.
// ---------------------------------------------------------------------------
template <int WM, int WN, bool BT, typename Epi>
__global__ __launch_bounds__(256)
void gemm_bf16(const unsigned short* __restrict__ A, long lda,
               const unsigned short* __restrict__ Bm, long ldb,
               int K, int zmod, long aSdiv, long aSmod, long bSdiv, long bSmod,
               Epi epi)
{
    constexpr int T   = 32 * WM * WN;
    constexpr int BM  = 32 * WM;
    constexpr int BN  = 32 * WN;
    constexpr int ASL = (BM * 4) / T;          // uint4 slots per thread (A)
    constexpr int BSL = (BN * 4) / T;          // uint4 slots per thread (B, BT)
    constexpr int BNN = (32 * BN / 8) / T;     // uint4 slots per thread (B, NN)

    __shared__ unsigned short As[2][BM][32];
    __shared__ unsigned short Bs[2][BN][32];

    const int bz = blockIdx.z;
    const unsigned short* Ag = A  + (long)(bz / zmod) * aSdiv + (long)(bz % zmod) * aSmod;
    const unsigned short* Bg = Bm + (long)(bz / zmod) * bSdiv + (long)(bz % zmod) * bSmod;

    const int m0 = blockIdx.y * BM;
    const int n0 = blockIdx.x * BN;
    const int t    = threadIdx.x;
    const int w    = t >> 5;
    const int lane = t & 31;
    const int waveM = w % WM;
    const int waveN = w / WM;
    const int lrow  = lane & 15;
    const int hl    = lane >> 4;

    auto stage = [&](int buf, int k0) {
#pragma unroll
        for (int it = 0; it < ASL; ++it) {
            int s = t + it * T, r = s >> 2, seg = (s & 3) * 8;
            cp16_g2l(&Ag[(long)(m0 + r) * lda + k0 + seg], &As[buf][r][seg]);
        }
        if (BT) {
#pragma unroll
            for (int it = 0; it < BSL; ++it) {
                int s = t + it * T, r = s >> 2, seg = (s & 3) * 8;
                cp16_g2l(&Bg[(long)(n0 + r) * ldb + k0 + seg], &Bs[buf][r][seg]);
            }
        } else {
#pragma unroll
            for (int it = 0; it < BNN; ++it) {
                int s = t + it * T;
                int kk = s / (BN / 8), nn = (s % (BN / 8)) * 8;
                uint4 d = *(const uint4*)&Bg[(long)(k0 + kk) * ldb + n0 + nn];
                const unsigned short* ds = (const unsigned short*)&d;
#pragma unroll
                for (int j = 0; j < 8; ++j) Bs[buf][nn + j][kk] = ds[j];
            }
        }
    };

    v8f acc[2][2] = {};
    const int nk = K >> 5;
    stage(0, 0);
    for (int ki = 0; ki < nk; ++ki) {
        const int buf = ki & 1;
        wait_async_lds();
        __syncthreads();
        if (ki + 1 < nk) stage(buf ^ 1, (ki + 1) * 32);

        Frag a[2], b[2];
#pragma unroll
        for (int i = 0; i < 2; ++i) {
            int r = waveM * 32 + i * 16 + lrow;
            a[i].q[0] = *(const uint4*)&As[buf][r][8 * hl];
            a[i].q[1] = *(const uint4*)&As[buf][r][16 + 8 * hl];
        }
#pragma unroll
        for (int j = 0; j < 2; ++j) {
            int r = waveN * 32 + j * 16 + lrow;
            b[j].q[0] = *(const uint4*)&Bs[buf][r][8 * hl];
            b[j].q[1] = *(const uint4*)&Bs[buf][r][16 + 8 * hl];
        }
#pragma unroll
        for (int i = 0; i < 2; ++i)
#pragma unroll
            for (int j = 0; j < 2; ++j)
                acc[i][j] = __builtin_amdgcn_wmma_f32_16x16x32_bf16(
                    false, a[i].v, false, b[j].v, (short)0, acc[i][j], false, false);
    }

#pragma unroll
    for (int i = 0; i < 2; ++i)
#pragma unroll
        for (int j = 0; j < 2; ++j) {
            int mB = m0 + waveM * 32 + i * 16 + 8 * hl;
            int n  = n0 + waveN * 32 + j * 16 + lrow;
#pragma unroll
            for (int r = 0; r < 8; ++r) epi(bz, mB + r, n, acc[i][j][r]);
        }
}

// ---------------------------------------------------------------------------
// Epilogue functors
// ---------------------------------------------------------------------------
struct EpBf16Plain {                 // plain bf16 store, unbatched
    unsigned short* C; long ldc;
    __device__ void operator()(int, int m, int n, float v) const {
        C[(long)m * ldc + n] = f2bf(v);
    }
};
struct EpScores {                    // scores fp32 [b*H+h][64][4096]
    float* C;
    __device__ void operator()(int bz, int m, int n, float v) const {
        C[((long)bz * 64 + m) * 4096 + n] = v;
    }
};
struct EpO {                         // o bf16 [b][l][h*64+d]  (head concat)
    unsigned short* C;
    __device__ void operator()(int bz, int m, int n, float v) const {
        int b = bz >> 4, h = bz & 15;
        C[((long)b * 64 + m) * 1024 + h * 64 + n] = f2bf(v);
    }
};
struct EpAttnT {                     // attn^T bf16 [b][e][l], + bo[e]
    unsigned short* C; const float* bo;
    __device__ void operator()(int, int m, int n, float v) const {
        int b = m >> 6, l = m & 63;
        C[((long)b * 1024 + n) * 64 + l] = f2bf(v + bo[n]);
    }
};
struct EpProj {                      // d_out = x + attn_proj + bproj[s]
    float* out; const float* x; const float* bproj;
    __device__ void operator()(int bz, int m, int n, float v) const {
        long idx = ((long)bz * 4096 + m) * 1024 + n;
        out[idx] = x[idx] + v + bproj[m];
    }
};
struct EpFFN2 {                      // d_out += ffn + b3[e]
    float* out; const float* b3;
    __device__ void operator()(int, int m, int n, float v) const {
        long idx = (long)m * 1024 + n;
        out[idx] += v + b3[n];
    }
};

// ---------------------------------------------------------------------------
// Fused SwiGLU dual-GEMM: g = swish(h2@W1+b1) * (h2@W2+b2), bf16 out [M][4096]
// W1T/W2T given transposed: [4096][1024] (BT path, async, double-buffered).
// ---------------------------------------------------------------------------
__global__ __launch_bounds__(256)
void ffn_gate(const unsigned short* __restrict__ A,
              const unsigned short* __restrict__ B1,
              const unsigned short* __restrict__ B2,
              const float* __restrict__ bias1, const float* __restrict__ bias2,
              unsigned short* __restrict__ G)
{
    constexpr int BM = 128, BN = 64, T = 256;
    __shared__ unsigned short As[2][BM][32];
    __shared__ unsigned short Bs1[2][BN][32];
    __shared__ unsigned short Bs2[2][BN][32];

    const int m0 = blockIdx.y * BM, n0 = blockIdx.x * BN;
    const int t = threadIdx.x, w = t >> 5, lane = t & 31;
    const int waveM = w % 4, waveN = w / 4;
    const int lrow = lane & 15, hl = lane >> 4;

    auto stage = [&](int buf, int k0) {
#pragma unroll
        for (int it = 0; it < 2; ++it) {            // A: 512 slots / 256 thr
            int s = t + it * T, r = s >> 2, seg = (s & 3) * 8;
            cp16_g2l(&A[(long)(m0 + r) * 1024 + k0 + seg], &As[buf][r][seg]);
        }
        {   int s = t, r = s >> 2, seg = (s & 3) * 8;   // B1/B2: 256 slots
            cp16_g2l(&B1[(long)(n0 + r) * 1024 + k0 + seg], &Bs1[buf][r][seg]);
            cp16_g2l(&B2[(long)(n0 + r) * 1024 + k0 + seg], &Bs2[buf][r][seg]);
        }
    };

    v8f acc1[2][2] = {}, acc2[2][2] = {};
    stage(0, 0);
    for (int ki = 0; ki < 32; ++ki) {               // K = 1024
        const int buf = ki & 1;
        wait_async_lds();
        __syncthreads();
        if (ki + 1 < 32) stage(buf ^ 1, (ki + 1) * 32);

        Frag a[2], f1[2], f2[2];
#pragma unroll
        for (int i = 0; i < 2; ++i) {
            int r = waveM * 32 + i * 16 + lrow;
            a[i].q[0] = *(const uint4*)&As[buf][r][8 * hl];
            a[i].q[1] = *(const uint4*)&As[buf][r][16 + 8 * hl];
        }
#pragma unroll
        for (int j = 0; j < 2; ++j) {
            int r = waveN * 32 + j * 16 + lrow;
            f1[j].q[0] = *(const uint4*)&Bs1[buf][r][8 * hl];
            f1[j].q[1] = *(const uint4*)&Bs1[buf][r][16 + 8 * hl];
            f2[j].q[0] = *(const uint4*)&Bs2[buf][r][8 * hl];
            f2[j].q[1] = *(const uint4*)&Bs2[buf][r][16 + 8 * hl];
        }
#pragma unroll
        for (int i = 0; i < 2; ++i)
#pragma unroll
            for (int j = 0; j < 2; ++j) {
                acc1[i][j] = __builtin_amdgcn_wmma_f32_16x16x32_bf16(
                    false, a[i].v, false, f1[j].v, (short)0, acc1[i][j], false, false);
                acc2[i][j] = __builtin_amdgcn_wmma_f32_16x16x32_bf16(
                    false, a[i].v, false, f2[j].v, (short)0, acc2[i][j], false, false);
            }
    }

#pragma unroll
    for (int i = 0; i < 2; ++i)
#pragma unroll
        for (int j = 0; j < 2; ++j) {
            int mB = m0 + waveM * 32 + i * 16 + 8 * hl;
            int n  = n0 + waveN * 32 + j * 16 + lrow;
#pragma unroll
            for (int r = 0; r < 8; ++r) {
                float av = acc1[i][j][r] + bias1[n];
                float gv = acc2[i][j][r] + bias2[n];
                float sw = av / (1.f + expf(-av));
                G[(long)(mB + r) * 4096 + n] = f2bf(sw * gv);
            }
        }
}

// ---------------------------------------------------------------------------
// LayerNorm over E=1024, one row per block (256 threads), bf16 output
// ---------------------------------------------------------------------------
__global__ __launch_bounds__(256)
void layernorm_bf16(const float* __restrict__ X, const float* __restrict__ g,
                    const float* __restrict__ b, unsigned short* __restrict__ O)
{
    long row = blockIdx.x;
    const float* x = X + row * 1024;
    float v[4], s1 = 0.f, s2 = 0.f;
#pragma unroll
    for (int i = 0; i < 4; ++i) {
        v[i] = x[threadIdx.x + 256 * i];
        s1 += v[i]; s2 += v[i] * v[i];
    }
    __shared__ float r1[8], r2[8];
    for (int o = 16; o > 0; o >>= 1) { s1 += __shfl_xor(s1, o, 32); s2 += __shfl_xor(s2, o, 32); }
    if ((threadIdx.x & 31) == 0) { r1[threadIdx.x >> 5] = s1; r2[threadIdx.x >> 5] = s2; }
    __syncthreads();
    if (threadIdx.x == 0) {
        float a = 0.f, c = 0.f;
        for (int i = 0; i < 8; ++i) { a += r1[i]; c += r2[i]; }
        r1[0] = a; r2[0] = c;
    }
    __syncthreads();
    float mean = r1[0] * (1.f / 1024.f);
    float var  = r2[0] * (1.f / 1024.f) - mean * mean;
    float rstd = rsqrtf(var + 1e-5f);
#pragma unroll
    for (int i = 0; i < 4; ++i) {
        int c = threadIdx.x + 256 * i;
        O[row * 1024 + c] = f2bf((v[i] - mean) * rstd * g[c] + b[c]);
    }
}

// ---------------------------------------------------------------------------
// Row softmax: rows = B*H*NL = 4096, cols = S = 4096, fp32 in, bf16 P out
// ---------------------------------------------------------------------------
__global__ __launch_bounds__(256)
void softmax_bf16(const float* __restrict__ Sc, unsigned short* __restrict__ P)
{
    long row = blockIdx.x;
    const float* x = Sc + row * 4096;
    __shared__ float red[8];

    float mx = -3.4e38f;
    for (int c = threadIdx.x; c < 4096; c += 256) mx = fmaxf(mx, x[c]);
    for (int o = 16; o > 0; o >>= 1) mx = fmaxf(mx, __shfl_xor(mx, o, 32));
    if ((threadIdx.x & 31) == 0) red[threadIdx.x >> 5] = mx;
    __syncthreads();
    if (threadIdx.x == 0) { float v = red[0]; for (int i = 1; i < 8; ++i) v = fmaxf(v, red[i]); red[0] = v; }
    __syncthreads();
    mx = red[0];
    __syncthreads();

    float sm = 0.f;
    for (int c = threadIdx.x; c < 4096; c += 256) sm += expf(x[c] - mx);
    for (int o = 16; o > 0; o >>= 1) sm += __shfl_xor(sm, o, 32);
    if ((threadIdx.x & 31) == 0) red[threadIdx.x >> 5] = sm;
    __syncthreads();
    if (threadIdx.x == 0) { float v = 0.f; for (int i = 0; i < 8; ++i) v += red[i]; red[0] = v; }
    __syncthreads();
    float inv = 1.f / red[0];
    for (int c = threadIdx.x; c < 4096; c += 256)
        P[row * 4096 + c] = f2bf(expf(x[c] - mx) * inv);
}

// ---------------------------------------------------------------------------
// q = (latents @ Wq) * HD^-0.5 per head, bf16 out [H][NL][HD]
// ---------------------------------------------------------------------------
__global__ __launch_bounds__(64)
void q_proj(const float* __restrict__ latents, const float* __restrict__ Wq,
            unsigned short* __restrict__ qbf)
{
    int h = blockIdx.x >> 6, l = blockIdx.x & 63;
    __shared__ float lat[256];
    for (int c = threadIdx.x; c < 256; c += 64) lat[c] = latents[((long)h * 64 + l) * 256 + c];
    __syncthreads();
    const float* W = Wq + (long)h * 256 * 64 + threadIdx.x;
    float s = 0.f;
    for (int c = 0; c < 256; ++c) s += lat[c] * W[(long)c * 64];
    qbf[((long)h * 64 + l) * 64 + threadIdx.x] = f2bf(s * 0.125f);   // 1/sqrt(64)
}

// ---------------------------------------------------------------------------
// Weight pack kernels (fp32 -> bf16, transposed for BT/async GEMM path)
// ---------------------------------------------------------------------------
// Generic transpose pack: in [R][C] fp32 -> out [C][R] bf16
__global__ void pack_bf16T(const float* __restrict__ in, unsigned short* __restrict__ out,
                           long R, long C)
{
    long i = (long)blockIdx.x * 256 + threadIdx.x;
    if (i >= R * C) return;
    long r = i / C, c = i - r * C;
    out[c * R + r] = f2bf(in[i]);
}
// Wk/Wv [H=16][E=1024][HD=64]  ->  [H*HD=1024][E=1024]  (Bt layout)
__global__ void pack_kvT(const float* __restrict__ W, unsigned short* __restrict__ out)
{
    long i = (long)blockIdx.x * 256 + threadIdx.x;
    if (i >= (long)16 * 1024 * 64) return;
    int d = i & 63, e = (int)((i >> 6) & 1023), h = (int)(i >> 16);
    out[((long)(h * 64 + d) << 10) + e] = f2bf(W[i]);
}

// ---------------------------------------------------------------------------
// Host launcher
// ---------------------------------------------------------------------------
extern "C" void kernel_launch(void* const* d_in, const int* in_sizes, int n_in,
                              void* d_out, int out_size, void* d_ws, size_t ws_size,
                              hipStream_t stream)
{
    (void)in_sizes; (void)n_in; (void)out_size; (void)ws_size;
    const float* x     = (const float*)d_in[0];
    const float* ln1g  = (const float*)d_in[1];
    const float* ln1b  = (const float*)d_in[2];
    const float* lat   = (const float*)d_in[3];
    const float* Wq    = (const float*)d_in[4];
    const float* Wk    = (const float*)d_in[5];
    const float* Wv    = (const float*)d_in[6];
    const float* Wo    = (const float*)d_in[7];
    const float* bo    = (const float*)d_in[8];
    const float* Wproj = (const float*)d_in[9];
    const float* bproj = (const float*)d_in[10];
    const float* ln2g  = (const float*)d_in[11];
    const float* ln2b  = (const float*)d_in[12];
    const float* W1    = (const float*)d_in[13];
    const float* b1    = (const float*)d_in[14];
    const float* W2    = (const float*)d_in[15];
    const float* b2    = (const float*)d_in[16];
    const float* W3    = (const float*)d_in[17];
    const float* b3    = (const float*)d_in[18];
    float* out = (float*)d_out;

    // ---- workspace carve (lifetime-overlapped) ----
    char* p = (char*)d_ws;
    auto carve = [&](size_t bytes) -> void* {
        void* r = (void*)p;
        p += (bytes + 255) & ~(size_t)255;
        return r;
    };
    unsigned short* hbuf  = (unsigned short*)carve(33554432);   // h (ln1) then h2 (ln2)
    unsigned short* kmat  = (unsigned short*)carve(33554432);   // K  [B*S][1024]
    unsigned short* vmat  = (unsigned short*)carve(33554432);   // V  [B*S][1024]; later W1T/W2T/W3T
    char*           gbuf  = (char*)carve(134217728);            // scores+P, later g [16384][4096]
    unsigned short* wkT   = (unsigned short*)carve(2097152);
    unsigned short* wvT   = (unsigned short*)carve(2097152);
    unsigned short* woT   = (unsigned short*)carve(2097152);
    unsigned short* wpT   = (unsigned short*)carve(524288);
    unsigned short* qbf   = (unsigned short*)carve(131072);
    unsigned short* obf   = (unsigned short*)carve(524288);
    unsigned short* attnT = (unsigned short*)carve(524288);

    float*          scores = (float*)gbuf;                      // 64 MB
    unsigned short* Pbuf   = (unsigned short*)(gbuf + 67108864);// 32 MB
    unsigned short* gmat   = (unsigned short*)gbuf;             // reused for SwiGLU output
    unsigned short* w1T    = vmat;                              // packed after V is dead
    unsigned short* w2T    = vmat + 4194304;
    unsigned short* w3T    = vmat + 8388608;

    // ---- phase 0: packs + LN1 + q ----
    pack_kvT<<<4096, 256, 0, stream>>>(Wk, wkT);
    pack_kvT<<<4096, 256, 0, stream>>>(Wv, wvT);
    pack_bf16T<<<4096, 256, 0, stream>>>(Wo, woT, 1024, 1024);     // woT[e_out][h*hd]
    pack_bf16T<<<1024, 256, 0, stream>>>(Wproj, wpT, 64, 4096);    // wpT[s][l]
    layernorm_bf16<<<16384, 256, 0, stream>>>(x, ln1g, ln1b, hbuf);
    q_proj<<<1024, 64, 0, stream>>>(lat, Wq, qbf);

    // ---- phase 1: K/V projections  [16384x1024] = h @ WkT^T (BT, async) ----
    gemm_bf16<4, 2, true><<<dim3(16, 128, 1), 256, 0, stream>>>(
        hbuf, 1024, wkT, 1024, 1024, 1, 0, 0, 0, 0, EpBf16Plain{kmat, 1024});
    gemm_bf16<4, 2, true><<<dim3(16, 128, 1), 256, 0, stream>>>(
        hbuf, 1024, wvT, 1024, 1024, 1, 0, 0, 0, 0, EpBf16Plain{vmat, 1024});

    // ---- phase 2: scores = q @ K^T per (b,h)  [64 x 4096], K=64 (BT) ----
    gemm_bf16<2, 4, true><<<dim3(32, 1, 64), 256, 0, stream>>>(
        qbf, 64, kmat, 1024, 64,
        /*zmod=*/16, /*aSdiv=*/0, /*aSmod=*/4096,
        /*bSdiv=*/4194304, /*bSmod=*/64, EpScores{scores});

    // ---- phase 3: softmax -> P (bf16) ----
    softmax_bf16<<<4096, 256, 0, stream>>>(scores, Pbuf);

    // ---- phase 4: o = P @ V per (b,h)  [64 x 64], K=4096 (NN) ----
    gemm_bf16<2, 2, false><<<dim3(1, 1, 64), 128, 0, stream>>>(
        Pbuf, 4096, vmat, 1024, 4096,
        /*zmod=*/16, /*aSdiv=*/4194304, /*aSmod=*/262144,
        /*bSdiv=*/4194304, /*bSmod=*/64, EpO{obf});

    // ---- phase 5: pack FFN weights (transposed) into dead V region ----
    pack_bf16T<<<16384, 256, 0, stream>>>(W1, w1T, 1024, 4096);    // [4096][1024]
    pack_bf16T<<<16384, 256, 0, stream>>>(W2, w2T, 1024, 4096);    // [4096][1024]
    pack_bf16T<<<16384, 256, 0, stream>>>(W3, w3T, 4096, 1024);    // [1024][4096]

    // ---- phase 6: attn = o @ Wo + bo  -> attn^T [b][e][l]  (BT) ----
    gemm_bf16<4, 2, true><<<dim3(16, 2, 1), 256, 0, stream>>>(
        obf, 1024, woT, 1024, 1024, 1, 0, 0, 0, 0, EpAttnT{attnT, bo});

    // ---- phase 7: out = x + WprojT@attnT + bproj  (BT) ----
    gemm_bf16<4, 2, true><<<dim3(16, 32, 4), 256, 0, stream>>>(
        wpT, 64, attnT, 64, 64,
        /*zmod=*/1, /*aSdiv=*/0, /*aSmod=*/0,
        /*bSdiv=*/65536, /*bSmod=*/0, EpProj{out, x, bproj});

    // ---- phase 8: LN2 (reuse hbuf) ----
    layernorm_bf16<<<16384, 256, 0, stream>>>(out, ln2g, ln2b, hbuf);

    // ---- phase 9: fused SwiGLU dual GEMM -> g [16384 x 4096] bf16 ----
    ffn_gate<<<dim3(64, 128, 1), 256, 0, stream>>>(hbuf, w1T, w2T, b1, b2, gmat);

    // ---- phase 10: out += g @ W3 + b3  (BT) ----
    gemm_bf16<4, 2, true><<<dim3(16, 128, 1), 256, 0, stream>>>(
        gmat, 4096, w3T, 4096, 4096, 1, 0, 0, 0, 0, EpFFN2{out, b3});
}
